// KernelizedAttention_4964982194187
// MI455X (gfx1250) — compile-verified, matrix operand order checked by reading, not run on hardware
//
#include <hip/hip_runtime.h>
#include <math.h>

// ---------------------------------------------------------------------------
// Kernelized (linear) causal attention for MI455X / gfx1250, wave32 + WMMA.
// N=2, L=2048, dim=512, H=8, d=64. Chunked linear attention, chunk C=64.
// All matmuls via v_wmma_f32_16x16x32_f16 (f16 operands, f32 accumulate).
// All WMMA fragments (A and B) are loaded with contiguous 16B loads: operands
// that need column access are stored *transposed* in LDS once per tile.
// ---------------------------------------------------------------------------

typedef __attribute__((ext_vector_type(16))) _Float16 v16h;
typedef __attribute__((ext_vector_type(8)))  _Float16 h8v;
typedef __attribute__((ext_vector_type(8)))  float    v8f;

#define WMMA_F16(a, b, c) \
  __builtin_amdgcn_wmma_f32_16x16x32_f16(false, (a), false, (b), (short)0, (c), false, false)

static __device__ __forceinline__ float softplus_f(float x) {
  // log(1 + exp(x)) = max(x,0) + log(1 + exp(-|x|)).
  // Result is stored as f16, so the native TRANS32 pipe (v_exp_f32/v_log_f32)
  // is more than accurate enough and co-executes with VALU work.
  float t = __expf(-fabsf(x));
  return fmaxf(x, 0.0f) + __logf(1.0f + t);
}

// Fragment from a row-major matrix M2 (rows 16B-aligned):
//  - as A-operand of a 16x16x32 WMMA for A = M2 (r0 = M offset, k0 = K offset)
//  - as B-operand for B = M2^T (r0 = N offset, k0 = K offset)
// CDNA5 16-bit layout: lane&15 -> row/col, lane bit4 selects the K-half,
// halves 0..7 = K[kb..kb+7], halves 8..15 = K[kb+16..kb+23].
static __device__ __forceinline__ v16h frag_rm(const _Float16* base, int r0,
                                               int k0, int ld, int lane) {
  int row = r0 + (lane & 15);
  int kb  = k0 + ((lane & 16) ? 8 : 0);
  const _Float16* p = base + row * ld + kb;
  h8v lo = *(const h8v*)p;
  h8v hi = *(const h8v*)(p + 16);
  return __builtin_shufflevector(lo, hi, 0, 1, 2, 3, 4, 5, 6, 7,
                                         8, 9, 10, 11, 12, 13, 14, 15);
}

// ---------------------------------------------------------------------------
// fp32 -> f16 elementwise conversion (inputs & weights; all L2-resident)
// ---------------------------------------------------------------------------
__global__ __launch_bounds__(256) void cvt_f32_f16(const float* __restrict__ src,
                                                   _Float16* __restrict__ dst,
                                                   int n) {
  int i = blockIdx.x * 256 + threadIdx.x;
  if (i < n) dst[i] = (_Float16)src[i];
}

// ---------------------------------------------------------------------------
// Projection GEMM: dst = act(X @ W^T), stored per-head: (N*H, L, d) in f16.
// M=4096, K=512, N=512. Register-blocked: each wave owns a 32x64 tile
// (2x4 accumulators); 8 waves -> 128x128 per workgroup. A/B fragments are
// contiguous 16B global loads (B operand = W^T comes straight from row-major
// W). 8 WMMAs per k-step reuse 2 A-frags x 4 B-frags.
// ---------------------------------------------------------------------------
__global__ __launch_bounds__(256) void proj_kernel(const _Float16* __restrict__ X,
                                                   const _Float16* __restrict__ W,
                                                   _Float16* __restrict__ dst,
                                                   int apply_sp) {
  int tid = threadIdx.x, wave = tid >> 5, lane = tid & 31;
  int Moff = blockIdx.y * 128 + (wave >> 1) * 32;
  int Noff = blockIdx.x * 128 + (wave & 1) * 64;

  v8f zero = {};
  v8f acc[2][4];
#pragma unroll
  for (int i = 0; i < 2; ++i)
#pragma unroll
    for (int j = 0; j < 4; ++j) acc[i][j] = zero;

#pragma unroll 2
  for (int k0 = 0; k0 < 512; k0 += 32) {
    v16h a0 = frag_rm(X, Moff, k0, 512, lane);
    v16h a1 = frag_rm(X, Moff + 16, k0, 512, lane);
    v16h b0 = frag_rm(W, Noff, k0, 512, lane);       // B[k][n] = W[n][k]
    v16h b1 = frag_rm(W, Noff + 16, k0, 512, lane);
    v16h b2 = frag_rm(W, Noff + 32, k0, 512, lane);
    v16h b3 = frag_rm(W, Noff + 48, k0, 512, lane);
    acc[0][0] = WMMA_F16(a0, b0, acc[0][0]);
    acc[0][1] = WMMA_F16(a0, b1, acc[0][1]);
    acc[0][2] = WMMA_F16(a0, b2, acc[0][2]);
    acc[0][3] = WMMA_F16(a0, b3, acc[0][3]);
    acc[1][0] = WMMA_F16(a1, b0, acc[1][0]);
    acc[1][1] = WMMA_F16(a1, b1, acc[1][1]);
    acc[1][2] = WMMA_F16(a1, b2, acc[1][2]);
    acc[1][3] = WMMA_F16(a1, b3, acc[1][3]);
  }

#pragma unroll
  for (int i = 0; i < 2; ++i) {
    int rbase = Moff + i * 16 + ((lane & 16) ? 8 : 0);
#pragma unroll
    for (int j = 0; j < 4; ++j) {
      int col = Noff + j * 16 + (lane & 15);
      int h = col >> 6, dd = col & 63;
#pragma unroll
      for (int r = 0; r < 8; ++r) {
        int row  = rbase + r;
        int bidx = row >> 11, l = row & 2047;  // 2048 rows per batch
        float val = acc[i][j][r];
        if (apply_sp) val = softplus_f(val);
        dst[(((size_t)(bidx * 8 + h)) * 2048 + l) * 64 + dd] = (_Float16)val;
      }
    }
  }
}

// ---------------------------------------------------------------------------
// Pass A: per-chunk state contributions. KV_c = PK_c^T @ V_c (64x64 f32),
// ksum_c = colsum(PK_c). Grid (chunk=32, nh=16), 512 independent WGs.
// PK and V tiles are stored TRANSPOSED in LDS so that both WMMA operands
// (A = PK^T, B = V) come from frag_rm (2x ds_load_b128 per fragment), and the
// colsum becomes a contiguous row sum.
// ---------------------------------------------------------------------------
__global__ __launch_bounds__(256) void kv_chunk_kernel(const _Float16* __restrict__ pk,
                                                       const _Float16* __restrict__ vv,
                                                       float* __restrict__ gKV,
                                                       float* __restrict__ gKsum) {
  __shared__ __align__(16) _Float16 sPKt[64 * 64];  // PK^T : [dd][s]
  __shared__ __align__(16) _Float16 sVt[64 * 64];   // V^T  : [dd][s]
  int tid = threadIdx.x, wave = tid >> 5, lane = tid & 31;
  int c = blockIdx.x, nh = blockIdx.y;
  size_t tileoff = ((size_t)nh * 2048 + (size_t)c * 64) * 64;

#pragma unroll
  for (int u = 0; u < 2; ++u) {
    int idx = tid + u * 256;      // h8v chunk index 0..511
    int s   = idx >> 3;           // source row (position in chunk)
    int dd0 = (idx & 7) * 8;      // first feature of this chunk
    h8v kv8 = ((const h8v*)(pk + tileoff))[idx];
    h8v vv8 = ((const h8v*)(vv + tileoff))[idx];
#pragma unroll
    for (int e = 0; e < 8; ++e) {
      sPKt[(dd0 + e) * 64 + s] = kv8[e];
      sVt[(dd0 + e) * 64 + s]  = vv8[e];
    }
  }
  __syncthreads();

  float* KV = gKV + ((size_t)nh * 32 + c) * 4096;
#pragma unroll
  for (int tt = 0; tt < 2; ++tt) {
    int tile = wave * 2 + tt, mi = tile >> 2, ni = tile & 3;
    v8f acc = {};
#pragma unroll
    for (int kk = 0; kk < 2; ++kk) {
      v16h a = frag_rm(sPKt, mi * 16, kk * 32, 64, lane);  // A = PK^T
      v16h b = frag_rm(sVt, ni * 16, kk * 32, 64, lane);   // B = (V^T)^T = V
      acc = WMMA_F16(a, b, acc);
    }
    int colD  = ni * 16 + (lane & 15);
    int rbase = mi * 16 + ((lane & 16) ? 8 : 0);
#pragma unroll
    for (int r = 0; r < 8; ++r) KV[(rbase + r) * 64 + colD] = acc[r];
  }
  if (tid < 64) {
    float s = 0.f;
    for (int ss = 0; ss < 64; ++ss) s += (float)sPKt[tid * 64 + ss];
    gKsum[((size_t)nh * 32 + c) * 64 + tid] = s;
  }
}

// ---------------------------------------------------------------------------
// Pass B: in-place exclusive prefix over the 32 chunks (per nh). Tiny.
// ---------------------------------------------------------------------------
__global__ __launch_bounds__(256) void scan_kernel(float* __restrict__ gKV,
                                                   float* __restrict__ gKsum) {
  int nh = blockIdx.x, tid = threadIdx.x;
  for (int e = tid * 16; e < tid * 16 + 16; ++e) {
    float run = 0.f;
    float* p = gKV + (size_t)nh * 32 * 4096 + e;
    for (int c = 0; c < 32; ++c) {
      float t = p[(size_t)c * 4096];
      p[(size_t)c * 4096] = run;
      run += t;
    }
  }
  if (tid < 64) {
    float run = 0.f;
    float* p = gKsum + (size_t)nh * 32 * 64 + tid;
    for (int c = 0; c < 32; ++c) {
      float t = p[c * 64];
      p[c * 64] = run;
      run += t;
    }
  }
}

// ---------------------------------------------------------------------------
// Pass C: per-chunk output. A = tril(PQ PK^T); out = (A V + PQ S_prev) / den,
// den_i = rowsum_i(A) + PQ_i . z_prev. Grid (chunk=32, nh=16).
// V and S_prev are stored transposed in LDS so every B fragment is frag_rm.
// ---------------------------------------------------------------------------
__global__ __launch_bounds__(256) void attn_chunk_kernel(const _Float16* __restrict__ pq,
                                                         const _Float16* __restrict__ pk,
                                                         const _Float16* __restrict__ vv,
                                                         const float* __restrict__ gKV,
                                                         const float* __restrict__ gKsum,
                                                         float* __restrict__ out) {
  __shared__ __align__(16) _Float16 sPQ[64 * 64];   // row-major [i][dd]
  __shared__ __align__(16) _Float16 sPK[64 * 64];   // row-major [s][dd]
  __shared__ __align__(16) _Float16 sVt[64 * 64];   // V^T  : [dd][s]
  __shared__ __align__(16) _Float16 sSt[64 * 64];   // S^T  : [DD][dd] (f16)
  __shared__ __align__(16) _Float16 sAh[64 * 64];   // masked scores, f16
  __shared__ float sA[64 * 64];
  __shared__ float zst[64];
  __shared__ float den[64];

  int tid = threadIdx.x, wave = tid >> 5, lane = tid & 31;
  int c = blockIdx.x, nh = blockIdx.y;
  int batch = nh >> 3, h = nh & 7;

  size_t tileoff = ((size_t)nh * 2048 + (size_t)c * 64) * 64;
  const h8v* gq = (const h8v*)(pq + tileoff);
  const h8v* gk = (const h8v*)(pk + tileoff);
  ((h8v*)sPQ)[tid] = gq[tid];  ((h8v*)sPQ)[tid + 256] = gq[tid + 256];
  ((h8v*)sPK)[tid] = gk[tid];  ((h8v*)sPK)[tid + 256] = gk[tid + 256];
#pragma unroll
  for (int u = 0; u < 2; ++u) {
    int idx = tid + u * 256;
    int s   = idx >> 3;
    int dd0 = (idx & 7) * 8;
    h8v vv8 = ((const h8v*)(vv + tileoff))[idx];
#pragma unroll
    for (int e = 0; e < 8; ++e) sVt[(dd0 + e) * 64 + s] = vv8[e];
  }
  const float* Sprev = gKV + ((size_t)nh * 32 + c) * 4096;
#pragma unroll
  for (int e = tid * 16; e < tid * 16 + 16; ++e) {
    int dd = e >> 6, DD = e & 63;
    sSt[DD * 64 + dd] = (_Float16)Sprev[e];    // transpose while converting
  }
  if (tid < 64) zst[tid] = gKsum[((size_t)nh * 32 + c) * 64 + tid];
  __syncthreads();

  // scores A = PQ @ PK^T, causal (inclusive) mask applied on write
#pragma unroll
  for (int tt = 0; tt < 2; ++tt) {
    int tile = wave * 2 + tt, mi = tile >> 2, ni = tile & 3;
    v8f acc = {};
#pragma unroll
    for (int kk = 0; kk < 2; ++kk) {
      v16h a = frag_rm(sPQ, mi * 16, kk * 32, 64, lane);
      v16h b = frag_rm(sPK, ni * 16, kk * 32, 64, lane);  // B = PK^T
      acc = WMMA_F16(a, b, acc);
    }
    int colj  = ni * 16 + (lane & 15);
    int rbase = mi * 16 + ((lane & 16) ? 8 : 0);
#pragma unroll
    for (int r = 0; r < 8; ++r) {
      int i = rbase + r;
      sA[i * 64 + colj] = (colj <= i) ? acc[r] : 0.0f;
    }
  }
  __syncthreads();

  // f16 scores + denominator
#pragma unroll
  for (int e = tid * 16; e < tid * 16 + 16; ++e) sAh[e] = (_Float16)sA[e];
  if (tid < 64) {
    float rs = 0.f;
    for (int j = 0; j < 64; ++j) rs += sA[tid * 64 + j];  // already masked
    float dz = 0.f;
    for (int dd = 0; dd < 64; ++dd) dz += (float)sPQ[tid * 64 + dd] * zst[dd];
    den[tid] = rs + dz;
  }
  __syncthreads();

  // out = (A_mask @ V + PQ @ S_prev) / den ; all fragments contiguous
#pragma unroll
  for (int tt = 0; tt < 2; ++tt) {
    int tile = wave * 2 + tt, mi = tile >> 2, ni = tile & 3;
    v8f acc = {};
#pragma unroll
    for (int kk = 0; kk < 2; ++kk)
      acc = WMMA_F16(frag_rm(sAh, mi * 16, kk * 32, 64, lane),
                     frag_rm(sVt, ni * 16, kk * 32, 64, lane), acc);
#pragma unroll
    for (int kk = 0; kk < 2; ++kk)
      acc = WMMA_F16(frag_rm(sPQ, mi * 16, kk * 32, 64, lane),
                     frag_rm(sSt, ni * 16, kk * 32, 64, lane), acc);
    int colD  = ni * 16 + (lane & 15);
    int rbase = mi * 16 + ((lane & 16) ? 8 : 0);
#pragma unroll
    for (int r = 0; r < 8; ++r) {
      int i = rbase + r;
      int l = c * 64 + i;
      out[((size_t)(batch * 2048 + l)) * 512 + h * 64 + colD] = acc[r] / den[i];
    }
  }
}

// ---------------------------------------------------------------------------
// Host-side launch. Workspace layout (30 MB):
//   qh 4MB | kh 4MB | Wqh .5MB | Wkh .5MB | Wvh .5MB |
//   pq 4MB | pk 4MB | v 4MB | KV prefix 8MB | ksum prefix .5MB
// ---------------------------------------------------------------------------
extern "C" void kernel_launch(void* const* d_in, const int* in_sizes, int n_in,
                              void* d_out, int out_size, void* d_ws, size_t ws_size,
                              hipStream_t stream) {
  (void)in_sizes; (void)n_in; (void)out_size; (void)ws_size;
  const float* query = (const float*)d_in[0];
  const float* key   = (const float*)d_in[1];
  const float* Wq    = (const float*)d_in[2];
  const float* Wk    = (const float*)d_in[3];
  const float* Wv    = (const float*)d_in[4];
  float* out = (float*)d_out;

  char* ws = (char*)d_ws;
  const size_t MB = 1024 * 1024, HK = 512 * 1024;
  _Float16* qh   = (_Float16*)(ws + 0);
  _Float16* kh   = (_Float16*)(ws + 4 * MB);
  _Float16* Wqh  = (_Float16*)(ws + 8 * MB);
  _Float16* Wkh  = (_Float16*)(ws + 8 * MB + HK);
  _Float16* Wvh  = (_Float16*)(ws + 9 * MB);
  _Float16* pqB  = (_Float16*)(ws + 9 * MB + HK);
  _Float16* pkB  = (_Float16*)(ws + 13 * MB + HK);
  _Float16* vvB  = (_Float16*)(ws + 17 * MB + HK);
  float* gKV     = (float*)(ws + 21 * MB + HK);
  float* gKsum   = (float*)(ws + 29 * MB + HK);

  const int nX = 2 * 2048 * 512;  // 2,097,152
  const int nW = 512 * 512;       // 262,144
  cvt_f32_f16<<<(nX + 255) / 256, 256, 0, stream>>>(query, qh, nX);
  cvt_f32_f16<<<(nX + 255) / 256, 256, 0, stream>>>(key, kh, nX);
  cvt_f32_f16<<<(nW + 255) / 256, 256, 0, stream>>>(Wq, Wqh, nW);
  cvt_f32_f16<<<(nW + 255) / 256, 256, 0, stream>>>(Wk, Wkh, nW);
  cvt_f32_f16<<<(nW + 255) / 256, 256, 0, stream>>>(Wv, Wvh, nW);

  dim3 pgrid(4, 32);  // N-tiles(128) x M-tiles(128)
  proj_kernel<<<pgrid, 256, 0, stream>>>(qh, Wqh, pqB, 1);
  proj_kernel<<<pgrid, 256, 0, stream>>>(kh, Wkh, pkB, 1);
  proj_kernel<<<pgrid, 256, 0, stream>>>(kh, Wvh, vvB, 0);

  dim3 agrid(32, 16);  // chunks x (batch*heads)
  kv_chunk_kernel<<<agrid, 256, 0, stream>>>(pkB, vvB, gKV, gKsum);
  scan_kernel<<<16, 256, 0, stream>>>(gKV, gKsum);
  attn_chunk_kernel<<<agrid, 256, 0, stream>>>(pqB, pkB, vvB, gKV, gKsum, out);
}
